// OpenAIMoeExperts_85890755985633
// MI455X (gfx1250) — compile-verified, matrix operand order checked by reading, not run on hardware
//
#include <hip/hip_runtime.h>

// ---------------------------------------------------------------------------
// Fused MoE experts (gate_up GEMM -> sigmoid-GLU -> down GEMM) for MI455X
// (gfx1250, wave32, WMMA).  Compute-bound (AI ~725 flop/byte) -> built around
// v_wmma_f32_16x16x32_bf16 (16384 FLOP/instr), fp32 accumulate.
//
//   prep kernels : fp32 -> bf16 convert of X; transpose+convert of W1/W2 into
//                  d_ws so the WMMA B-operand is K-contiguous in memory.
//   main kernel  : one WG = (expert, 64-token tile). X tile + activation tile
//                  resident in LDS (CDNA5: 320KB/WG).  B slabs stream via
//                  GLOBAL_LOAD_ASYNC_TO_LDS_B128 (ASYNCcnt) into
//                  TRIPLE-buffered LDS, issued 2 k-steps ahead; in-order async
//                  completion lets s_wait_asynccnt 4 keep the newest slab in
//                  flight across the barrier.  Zero staging VGPRs, no spills.
// ---------------------------------------------------------------------------

#define ALPHA 1.702f

constexpr int E_ = 32;
constexpr int T_ = 4096;
constexpr int H_ = 768;   // hidden
constexpr int I_ = 768;   // intermediate

constexpr int TM  = 64;   // token tile per workgroup
constexpr int BN  = 128;  // N columns per pass
constexpr int BK  = 32;   // K slab per step (== WMMA K)
constexpr int BLD = 40;   // padded LDS row (elems) for B tiles

constexpr int NKS = H_ / BK;  // 24 k-steps (H_ == I_)

// LDS layout (elements of unsigned short)
constexpr int OFF_XL = 0;                 // [TM][H]   96 KB
constexpr int OFF_AL = TM * H_;           // [TM][I]   96 KB
constexpr int BSZ    = BN * BLD;          // 5120 elems = 10 KB
constexpr int OFF_B0 = OFF_AL + TM * I_;  // 3 buffers x {gate,up}
constexpr int LDS_ELEMS = OFF_B0 + 6 * BSZ;   // 252 KB total

typedef __bf16 bf16x16 __attribute__((ext_vector_type(16)));
typedef float  f32x8   __attribute__((ext_vector_type(8)));

__device__ __forceinline__ unsigned short f2bf(float f) {
  unsigned int u = __float_as_uint(f);
  u += 0x7fffu + ((u >> 16) & 1u);          // round-to-nearest-even
  return (unsigned short)(u >> 16);
}

struct alignas(8) US4 { unsigned short x, y, z, w; };

// ---------------------------------------------------------------------------
// Prep kernel 1: elementwise fp32 -> bf16 (for hidden_states)
// ---------------------------------------------------------------------------
__global__ void cvt_bf16_kernel(const float* __restrict__ in,
                                unsigned short* __restrict__ out, long n) {
  const long i = ((long)blockIdx.x * blockDim.x + threadIdx.x) * 8;
  if (i + 8 <= n) {
    const float4 a = *reinterpret_cast<const float4*>(in + i);
    const float4 b = *reinterpret_cast<const float4*>(in + i + 4);
    US4 o0, o1;
    o0.x = f2bf(a.x); o0.y = f2bf(a.y); o0.z = f2bf(a.z); o0.w = f2bf(a.w);
    o1.x = f2bf(b.x); o1.y = f2bf(b.y); o1.z = f2bf(b.z); o1.w = f2bf(b.w);
    *reinterpret_cast<US4*>(out + i)     = o0;
    *reinterpret_cast<US4*>(out + i + 4) = o1;
  }
}

// ---------------------------------------------------------------------------
// Prep kernel 2: per-slab transpose + convert:  in [R][C] fp32 -> out [C][R]
// bf16.  blockIdx.z selects the expert slab; 32x32 tiles via LDS.
// ---------------------------------------------------------------------------
__global__ void transpose_cvt_kernel(const float* __restrict__ in,
                                     unsigned short* __restrict__ out,
                                     int R, int C) {
  __shared__ unsigned short tile[32][33];
  const size_t slab = (size_t)R * C;
  const float* ip = in + (size_t)blockIdx.z * slab;
  unsigned short* op = out + (size_t)blockIdx.z * slab;
  const int c0 = blockIdx.x * 32;
  const int r0 = blockIdx.y * 32;
  {
    const int r = threadIdx.x >> 3;
    const int c = (threadIdx.x & 7) << 2;
    const float4 v =
        *reinterpret_cast<const float4*>(ip + (size_t)(r0 + r) * C + c0 + c);
    tile[r][c + 0] = f2bf(v.x);
    tile[r][c + 1] = f2bf(v.y);
    tile[r][c + 2] = f2bf(v.z);
    tile[r][c + 3] = f2bf(v.w);
  }
  __syncthreads();
  {
    const int c = threadIdx.x >> 3;        // out row  (original col)
    const int r = (threadIdx.x & 7) << 2;  // out col4 (original row)
    US4 o;
    o.x = tile[r + 0][c]; o.y = tile[r + 1][c];
    o.z = tile[r + 2][c]; o.w = tile[r + 3][c];
    *reinterpret_cast<US4*>(op + (size_t)(c0 + c) * R + r0 + r) = o;
  }
}

// ---------------------------------------------------------------------------
// Main fused kernel helpers
// ---------------------------------------------------------------------------

// CDNA5 async global->LDS copy, 16B per lane, tracked by ASYNCcnt
// (ISA 10.7 / 15.18.3: GLOBAL_LOAD_ASYNC_TO_LDS_B128).
__device__ __forceinline__ void async_b128(unsigned lds_byte,
                                           const void* gaddr) {
  asm volatile("global_load_async_to_lds_b128 %0, %1, off"
               :: "v"(lds_byte), "v"(gaddr) : "memory");
}
// Async loads complete in order (per wave), so waiting for <=N outstanding
// guarantees all but the newest N have landed in LDS.
template <int N>
__device__ __forceinline__ void wait_async() {
  asm volatile("s_wait_asynccnt %0" :: "i"(N) : "memory");
}

// Stage a BN x BK bf16 slab (K-contiguous rows, row stride ldk elems) into an
// LDS tile at byte base `dstB` (rows of BLD elems): 512 async b128 chunks,
// 2 per thread, zero data VGPRs.
__device__ __forceinline__ void bstage(const unsigned short* __restrict__ WT,
                                       int ldk, unsigned dstB) {
  const int n  = threadIdx.x >> 2;          // 0..63
  const int kc = (threadIdx.x & 3) << 3;    // 0,8,16,24
  async_b128(dstB + (unsigned)((n * BLD + kc) * 2),
             WT + (size_t)n * ldk + kc);
  async_b128(dstB + (unsigned)(((n + 64) * BLD + kc) * 2),
             WT + (size_t)(n + 64) * ldk + kc);
}

// WMMA fragment from an LDS tile of 16 rows x 32 K-contiguous bf16.
// ISA 7.12.2 (16-bit 16x32): lane L holds row (L&15); elems 0..7 = K h*8..,
// elems 8..15 = K 16+h*8.. (h = L>>4) -> two ds_load_b128.
__device__ __forceinline__ bf16x16 ld_frag(const unsigned short* S, int off,
                                           int ld) {
  const int lane = threadIdx.x & 31;
  const unsigned short* q = S + off + (lane & 15) * ld + (lane >> 4) * 8;
  union { uint4 q2[2]; bf16x16 v; } cv;
  cv.q2[0] = *reinterpret_cast<const uint4*>(q);
  cv.q2[1] = *reinterpret_cast<const uint4*>(q + 16);
  return cv.v;
}

__device__ __forceinline__ f32x8 wmma_bf16(bf16x16 a, bf16x16 b, f32x8 c) {
  return __builtin_amdgcn_wmma_f32_16x16x32_bf16(false, a, false, b,
                                                 (short)0, c, false, false);
}

__launch_bounds__(256, 1)
__global__ void moe_fused_kernel(const unsigned short* __restrict__ XT,  // [T,H] bf16
                                 const unsigned short* __restrict__ W1T, // [E,2I,H] bf16
                                 const unsigned short* __restrict__ W2T, // [E,H,I] bf16
                                 const float* __restrict__ B1,           // [E,2I]
                                 const float* __restrict__ B2,           // [E,H]
                                 float* __restrict__ out) {              // [E*T,H]
  extern __shared__ unsigned short smem[];
  unsigned short* const S = smem;

  // Byte base of dynamic LDS (no static LDS in this kernel).
  const unsigned dynB = __builtin_amdgcn_groupstaticsize();
  const unsigned xlB = dynB + OFF_XL * 2;
  const unsigned b0B = dynB + OFF_B0 * 2;   // buf c gate: +c*4*BSZ B; up: +2*BSZ B

  const int e  = blockIdx.y;
  const int t0 = blockIdx.x * TM;
  const int wid  = threadIdx.x >> 5;
  const int wm   = wid >> 2;          // 0..1
  const int wn   = wid & 3;           // 0..3
  const int lane = threadIdx.x & 31;
  const int nl   = lane & 15;
  const int half = lane >> 4;

  const unsigned short* W1e = W1T + (size_t)e * (2 * I_) * H_; // row f, K=h contig
  const unsigned short* W2e = W2T + (size_t)e * H_ * I_;       // row h, K=i contig

  // ---- stage X token tile into LDS (async b128 copies) --------------------
  for (int idx = threadIdx.x; idx < TM * (H_ / 8); idx += 256) {
    const int r  = idx / (H_ / 8);
    const int c8 = (idx % (H_ / 8)) * 8;
    async_b128(xlB + (unsigned)((r * H_ + c8) * 2),
               XT + (size_t)(t0 + r) * H_ + c8);
  }

  const f32x8 vzero = {0.f, 0.f, 0.f, 0.f, 0.f, 0.f, 0.f, 0.f};

  // ======================= Stage 1: gate_up + GLU ==========================
  for (int p = 0; p < I_ / BN; ++p) {
    const int n0 = p * BN;
    const unsigned short* Wg = W1e + (size_t)n0 * H_;        // gate rows
    const unsigned short* Wu = W1e + (size_t)(I_ + n0) * H_; // up rows

    f32x8 accG[2][2], accU[2][2];
#pragma unroll
    for (int a = 0; a < 2; ++a)
#pragma unroll
      for (int b = 0; b < 2; ++b) { accG[a][b] = vzero; accU[a][b] = vzero; }

    // prologue: slabs 0 and 1 into buffers 0 and 1 (also flushes X asyncs)
    bstage(Wg, H_, b0B);
    bstage(Wu, H_, b0B + 2 * BSZ);
    bstage(Wg + BK, H_, b0B + 4 * BSZ);
    bstage(Wu + BK, H_, b0B + 4 * BSZ + 2 * BSZ);
    wait_async<0>();
    __syncthreads();

#pragma unroll 3
    for (int ks = 0; ks < NKS; ++ks) {
      const int cur = ks % 3;                      // constant after unroll
      const int bg  = OFF_B0 + cur * (2 * BSZ);    // elem offsets (frag reads)
      const int bu  = bg + BSZ;
      const int k0  = ks * BK;
      const bool pre = (ks + 2 < NKS);

      if (pre) {                                   // fire slab ks+2
        const unsigned dstB = b0B + (unsigned)(((ks + 2) % 3) * (4 * BSZ));
        bstage(Wg + (size_t)(ks + 2) * BK, H_, dstB);
        bstage(Wu + (size_t)(ks + 2) * BK, H_, dstB + 2 * BSZ);
      }

      const bf16x16 a0 = ld_frag(S, OFF_XL + (wm * 32 + 0)  * H_ + k0, H_);
      const bf16x16 a1 = ld_frag(S, OFF_XL + (wm * 32 + 16) * H_ + k0, H_);
      {
        const bf16x16 g0 = ld_frag(S, bg + (wn * 32 + 0)  * BLD, BLD);
        const bf16x16 g1 = ld_frag(S, bg + (wn * 32 + 16) * BLD, BLD);
        accG[0][0] = wmma_bf16(a0, g0, accG[0][0]);
        accG[1][0] = wmma_bf16(a1, g0, accG[1][0]);
        accG[0][1] = wmma_bf16(a0, g1, accG[0][1]);
        accG[1][1] = wmma_bf16(a1, g1, accG[1][1]);
      }
      {
        const bf16x16 u0 = ld_frag(S, bu + (wn * 32 + 0)  * BLD, BLD);
        const bf16x16 u1 = ld_frag(S, bu + (wn * 32 + 16) * BLD, BLD);
        accU[0][0] = wmma_bf16(a0, u0, accU[0][0]);
        accU[1][0] = wmma_bf16(a1, u0, accU[1][0]);
        accU[0][1] = wmma_bf16(a0, u1, accU[0][1]);
        accU[1][1] = wmma_bf16(a1, u1, accU[1][1]);
      }
      // slab ks+1 (issued last iter) must have landed; slab ks+2 (the 4
      // asyncs just issued) may stay in flight -- in-order completion.
      if (pre) wait_async<4>(); else wait_async<0>();
      __syncthreads();
    }

    // --- epilogue: bias + sigmoid-GLU -> bf16 act tile in LDS -------------
    // C/D layout: VGPR j, lane L -> row j + 8*(L>>4), col (L&15).
#pragma unroll
    for (int tn = 0; tn < 2; ++tn) {
      const int col = n0 + wn * 32 + tn * 16 + nl;          // 0..I-1
      const float bg = B1[(size_t)e * (2 * I_) + col];
      const float bu = B1[(size_t)e * (2 * I_) + I_ + col];
#pragma unroll
      for (int tm = 0; tm < 2; ++tm) {
#pragma unroll
        for (int j = 0; j < 8; ++j) {
          const float g = accG[tm][tn][j] + bg;
          const float u = accU[tm][tn][j] + bu;
          const float glu = g * (1.0f / (1.0f + __expf(-ALPHA * g)));
          const float a = (u + 1.0f) * glu;
          const int row = wm * 32 + tm * 16 + half * 8 + j;
          S[OFF_AL + row * I_ + col] = f2bf(a);
        }
      }
    }
  }

  // ========================= Stage 2: down proj ============================
  for (int p = 0; p < H_ / BN; ++p) {
    const int n0 = p * BN;
    const unsigned short* Wd = W2e + (size_t)n0 * I_;

    f32x8 acc[2][2];
#pragma unroll
    for (int a = 0; a < 2; ++a)
#pragma unroll
      for (int b = 0; b < 2; ++b) acc[a][b] = vzero;

    bstage(Wd, I_, b0B);
    bstage(Wd + BK, I_, b0B + 4 * BSZ);
    wait_async<0>();
    __syncthreads();   // also orders stage-1 act writes before first read

#pragma unroll 3
    for (int ks = 0; ks < NKS; ++ks) {
      const int cur = ks % 3;
      const int bb  = OFF_B0 + cur * (2 * BSZ);
      const int k0  = ks * BK;
      const bool pre = (ks + 2 < NKS);

      if (pre)
        bstage(Wd + (size_t)(ks + 2) * BK, I_,
               b0B + (unsigned)(((ks + 2) % 3) * (4 * BSZ)));

      const bf16x16 a0 = ld_frag(S, OFF_AL + (wm * 32 + 0)  * I_ + k0, I_);
      const bf16x16 a1 = ld_frag(S, OFF_AL + (wm * 32 + 16) * I_ + k0, I_);
      const bf16x16 b0 = ld_frag(S, bb + (wn * 32 + 0)  * BLD, BLD);
      const bf16x16 b1 = ld_frag(S, bb + (wn * 32 + 16) * BLD, BLD);

      acc[0][0] = wmma_bf16(a0, b0, acc[0][0]);
      acc[1][0] = wmma_bf16(a1, b0, acc[1][0]);
      acc[0][1] = wmma_bf16(a0, b1, acc[0][1]);
      acc[1][1] = wmma_bf16(a1, b1, acc[1][1]);

      if (pre) wait_async<2>(); else wait_async<0>();
      __syncthreads();
    }

    // --- epilogue: down bias + fp32 store ---------------------------------
#pragma unroll
    for (int tn = 0; tn < 2; ++tn) {
      const int col = n0 + wn * 32 + tn * 16 + nl;          // 0..H-1
      const float bb = B2[(size_t)e * H_ + col];
#pragma unroll
      for (int tm = 0; tm < 2; ++tm) {
#pragma unroll
        for (int j = 0; j < 8; ++j) {
          const int row = t0 + wm * 32 + tm * 16 + half * 8 + j;
          out[((size_t)e * T_ + row) * H_ + col] = acc[tm][tn][j] + bb;
        }
      }
    }
  }
}

// ---------------------------------------------------------------------------
extern "C" void kernel_launch(void* const* d_in, const int* in_sizes, int n_in,
                              void* d_out, int out_size, void* d_ws,
                              size_t ws_size, hipStream_t stream) {
  (void)in_sizes; (void)n_in; (void)out_size; (void)ws_size;

  const float* X  = (const float*)d_in[0];  // [T,H]
  const float* W1 = (const float*)d_in[1];  // [E,H,2I]
  const float* B1 = (const float*)d_in[2];  // [E,2I]
  const float* W2 = (const float*)d_in[3];  // [E,I,H]
  const float* B2 = (const float*)d_in[4];  // [E,H]
  float* out = (float*)d_out;               // [E*T,H]

  // workspace layout (bf16): XT | W1T | W2T  = 119.5 MB
  unsigned short* XT  = (unsigned short*)d_ws;
  unsigned short* W1T = XT + (size_t)T_ * H_;
  unsigned short* W2T = W1T + (size_t)E_ * H_ * (2 * I_);

  // prep: ~340 MB of pure streaming -> ~15 us at 23.3 TB/s (noise vs compute)
  cvt_bf16_kernel<<<dim3((T_ * H_) / (256 * 8)), dim3(256), 0, stream>>>(
      X, XT, (long)T_ * H_);
  transpose_cvt_kernel<<<dim3((2 * I_) / 32, H_ / 32, E_), dim3(256), 0,
                         stream>>>(W1, W1T, H_, 2 * I_);
  transpose_cvt_kernel<<<dim3(H_ / 32, I_ / 32, E_), dim3(256), 0, stream>>>(
      W2, W2T, I_, H_);

  const size_t shmem = (size_t)LDS_ELEMS * sizeof(unsigned short); // 252 KB
  (void)hipFuncSetAttribute(reinterpret_cast<const void*>(moe_fused_kernel),
                            hipFuncAttributeMaxDynamicSharedMemorySize,
                            (int)shmem);

  dim3 grid(T_ / TM, E_);   // x fastest -> same-expert blocks adjacent (L2)
  dim3 block(256);          // 8 waves (wave32)
  moe_fused_kernel<<<grid, block, shmem, stream>>>(XT, W1T, W2T, B1, B2, out);
}
